// SelfAttention_7404523619240
// MI455X (gfx1250) — compile-verified, matrix operand order checked by reading, not run on hardware
//
#include <hip/hip_runtime.h>
#include <hip/hip_bf16.h>
#include <math.h>

// ---------------------------------------------------------------------------
// Causal self-attention for MI455X (gfx1250, wave32, WMMA f16->f32).
//   HIDDEN=1024, DK=DV=64, B=4, T=4096
// Pipeline:
//   k0: convert weights fp32 -> f16 into workspace
//   k1: fused QKV projection (WMMA, K=1024); q pre-scaled by 1/sqrt(64);
//       V stored pre-transposed (v_t[b][d][t]) for transpose-free attention
//   k2: flash attention, double-buffered KV chunks staged with
//       GLOBAL_LOAD_ASYNC_TO_LDS_B128 (ASYNCcnt pipeline)
//   k3: output projection (WMMA, K=64, N=1024) + bias, fp32 out
// ---------------------------------------------------------------------------

typedef __attribute__((ext_vector_type(16))) _Float16 v16h;
typedef __attribute__((ext_vector_type(8)))  _Float16 v8h;
typedef __attribute__((ext_vector_type(4)))  _Float16 v4h;
typedef __attribute__((ext_vector_type(8)))  float    v8f;
typedef __attribute__((ext_vector_type(4)))  float    v4f;
typedef __attribute__((ext_vector_type(4)))  int      v4i;

#define HID   1024
#define DK    64
#define SEQ   4096
#define NBAT  4
#define NROWS (NBAT * SEQ)          // 16384 token rows

// ---- gfx1250 async copy to LDS (guarded; fallback = vector copy) ----------
#if __has_builtin(__builtin_amdgcn_global_load_async_to_lds_b128)
#define HAVE_ASYNC 1
typedef __attribute__((address_space(1))) v4i as1_v4i;
typedef __attribute__((address_space(3))) v4i as3_v4i;
#else
#define HAVE_ASYNC 0
#endif

static __device__ __forceinline__ void copy16B(const _Float16* src, _Float16* dst) {
#if HAVE_ASYNC
  // prototype: (int4 AS1* src, int4 AS3* dst, imm offset, imm cpol)
  __builtin_amdgcn_global_load_async_to_lds_b128(
      (as1_v4i*)(uintptr_t)src,
      (as3_v4i*)(unsigned int)(uintptr_t)dst,   // low 32 bits = LDS offset
      0, 0);
#else
  *(v8h*)dst = *(const v8h*)src;
#endif
}

static __device__ __forceinline__ void wait_async_zero() {
#if HAVE_ASYNC
#if __has_builtin(__builtin_amdgcn_s_wait_asynccnt)
  __builtin_amdgcn_s_wait_asynccnt(0);
#else
  asm volatile("s_wait_asynccnt 0x0" ::: "memory");
#endif
#endif
}

// ---- fragment helpers (CDNA5 16x16x32 f16 WMMA layouts, wave32) -----------
// A-matrix 16x32: lane = hid*16 + r ; row M = r ;
//   halves e=0..7  -> K = hid*8 + e ; halves e=8..15 -> K = 16 + hid*8 + (e-8)
// B-matrix 32x16: same pattern with lane%16 = column N, reading src[N][K].
static __device__ __forceinline__ v16h frag_load(const _Float16* src, int ld) {
  const int lane = threadIdx.x & 31;
  const int hid  = lane >> 4;
  const int r    = lane & 15;
  const _Float16* p = src + (size_t)r * ld + hid * 8;
  v16h f;
#pragma unroll
  for (int e = 0; e < 8; ++e) f[e] = p[e];
#pragma unroll
  for (int e = 0; e < 8; ++e) f[8 + e] = p[16 + e];
  return f;
}

static __device__ __forceinline__ v8f wmma16(v16h a, v16h b, v8f c) {
  return __builtin_amdgcn_wmma_f32_16x16x32_f16(false, a, false, b,
                                                (short)0, c, false, false);
}

// ---------------------------------------------------------------------------
// k0: fp32 -> f16 weight conversion.
// ---------------------------------------------------------------------------
__global__ void k0_convert_w(const float* __restrict__ Wq,
                             const float* __restrict__ Wk,
                             const float* __restrict__ Wv,
                             const float* __restrict__ Wo,
                             _Float16* __restrict__ Wqkv_h,
                             _Float16* __restrict__ Wo_h) {
  const int idx = blockIdx.x * blockDim.x + threadIdx.x;
  const int WSZ = DK * HID;                 // 65536
  if (idx < WSZ) {
    Wqkv_h[idx] = (_Float16)Wq[idx];
  } else if (idx < 2 * WSZ) {
    Wqkv_h[idx] = (_Float16)Wk[idx - WSZ];
  } else if (idx < 3 * WSZ) {
    Wqkv_h[idx] = (_Float16)Wv[idx - 2 * WSZ];
  } else if (idx < 3 * WSZ + HID * DK) {
    Wo_h[idx - 3 * WSZ] = (_Float16)Wo[idx - 3 * WSZ];
  }
}

// ---------------------------------------------------------------------------
// k1: fused QKV projection.  grid.x = NROWS/16; 384 threads = 12 waves.
//   Wave w owns one 16-col tile of fused [q|k|v]; y tile staged in LDS.
//   V written transposed: v_t[b][d][t].
// ---------------------------------------------------------------------------
#define Y_LD (HID + 8)

__global__ void k1_qkv(const float* __restrict__ y,
                       const _Float16* __restrict__ Wqkv_h,
                       const float* __restrict__ bq,
                       const float* __restrict__ bk,
                       const float* __restrict__ bv,
                       _Float16* __restrict__ q_h,
                       _Float16* __restrict__ k_h,
                       _Float16* __restrict__ v_t) {
  __shared__ __align__(16) _Float16 ytile[16 * Y_LD];

  const int tid  = threadIdx.x;
  const int row0 = blockIdx.x * 16;

  // vectorized fp32->f16 stage of the 16x1024 y tile
  for (int i = tid * 4; i < 16 * HID; i += 384 * 4) {
    const int r = i >> 10, c = i & (HID - 1);
    const v4f yv = *(const v4f*)&y[(size_t)(row0 + r) * HID + c];
    v4h hv;
    hv[0] = (_Float16)yv[0]; hv[1] = (_Float16)yv[1];
    hv[2] = (_Float16)yv[2]; hv[3] = (_Float16)yv[3];
    *(v4h*)&ytile[r * Y_LD + c] = hv;
  }
  __syncthreads();

  const int w    = tid >> 5;                // 0..11
  const int lane = tid & 31;
  const int hid  = lane >> 4;
  const int n    = lane & 15;

  v8f acc = {};
#pragma unroll 4
  for (int kk = 0; kk < HID; kk += 32) {
    v16h a = frag_load(ytile + kk, Y_LD);
    v16h b = frag_load(Wqkv_h + (size_t)(w * 16) * HID + kk, HID);
    acc = wmma16(a, b, acc);
  }

  const int sel  = w >> 2;                  // 0=q 1=k 2=v
  const int col  = (w & 3) * 16 + n;        // 0..63 within q/k/v
  const float* bias = (sel == 0) ? bq : (sel == 1) ? bk : bv;
  const float scale = (sel == 0) ? 0.125f : 1.0f;   // q / sqrt(64)
  const float bval  = bias[col];
  _Float16* dst = (sel == 0) ? q_h : k_h;

#pragma unroll
  for (int j = 0; j < 8; ++j) {
    const int m = j + 8 * hid;
    const float val = (acc[j] + bval) * scale;
    if (sel == 2) {
      const int grow = row0 + m;
      v_t[(size_t)(grow >> 12) * (DK * SEQ) + (size_t)col * SEQ +
          (grow & (SEQ - 1))] = (_Float16)val;
    } else {
      dst[(size_t)(row0 + m) * DK + col] = (_Float16)val;
    }
  }
}

// ---------------------------------------------------------------------------
// k2: flash attention (causal), double-buffered async KV staging.
//   grid.x = NBAT * (SEQ/128); 256 threads = 8 waves; wave owns 16 q rows.
// ---------------------------------------------------------------------------
#define K_LD  88     // 64 + 24 pad halves: 176B stride (16B aligned, no bank conflict)
#define VT_LD 40     // 32 + 8 pad halves:  80B stride  (16B aligned, no bank conflict)
#define QTILE 128
#define KCH_SZ (32 * K_LD)
#define VCH_SZ (DK * VT_LD)

__global__ void k2_attn(const _Float16* __restrict__ q_h,
                        const _Float16* __restrict__ k_h,
                        const _Float16* __restrict__ v_t,
                        _Float16* __restrict__ ctx_h) {
  __shared__ __align__(16) _Float16 Klds[2 * KCH_SZ];   // K chunk [kv 32][d 64]
  __shared__ __align__(16) _Float16 Vlds[2 * VCH_SZ];   // V^T chunk [d 64][kv 32]
  __shared__ __align__(16) _Float16 Plds[8 * 16 * 32];  // per-wave P staging

  const int tid  = threadIdx.x;
  const int w    = tid >> 5;
  const int lane = tid & 31;
  const int hid  = lane >> 4;
  const int n    = lane & 15;

  const int qtiles = SEQ / QTILE;           // 32
  const int b      = blockIdx.x / qtiles;
  const int qbase  = (blockIdx.x % qtiles) * QTILE;
  const size_t bo64 = (size_t)b * SEQ * DK; // batch offset, [t][d] layouts
  const _Float16* kg  = k_h + bo64;
  const _Float16* vtg = v_t + bo64;         // [d][t] layout

  // per-thread staging coordinates (each thread moves exactly 2 x 16B / chunk)
  const int rK = tid >> 3, sK = tid & 7;    // K: row kv, 8-half segment
  const int dV = tid >> 2, sV = tid & 3;    // V^T: row d, 8-half segment

  // persistent Q fragments for this wave's 16 rows (already scaled)
  const _Float16* qp = q_h + bo64 + (size_t)(qbase + w * 16) * DK;
  const v16h aq0 = frag_load(qp, DK);
  const v16h aq1 = frag_load(qp + 32, DK);

  v8f   acc[4] = {v8f{}, v8f{}, v8f{}, v8f{}};
  float m_run[8], l_run[8];
#pragma unroll
  for (int j = 0; j < 8; ++j) { m_run[j] = -3.0e38f; l_run[j] = 0.0f; }

  _Float16* myP = Plds + w * (16 * 32);
  const int limit = qbase + QTILE;

  // stage chunk 0 into buffer 0
  copy16B(kg  + (size_t)(0 + rK) * DK + sK * 8, Klds + rK * K_LD  + sK * 8);
  copy16B(vtg + (size_t)dV * SEQ + 0 + sV * 8,  Vlds + dV * VT_LD + sV * 8);

  int cur = 0;
  for (int kvb = 0; kvb < limit; kvb += 32) {
    wait_async_zero();      // my async copies into buf[cur] done
    __syncthreads();        // everyone's copies visible; prev compute done

    // prefetch next chunk into the other buffer (overlaps with WMMAs below)
    if (kvb + 32 < limit) {
      const int nxt = kvb + 32;
      copy16B(kg  + (size_t)(nxt + rK) * DK + sK * 8,
              Klds + (cur ^ 1) * KCH_SZ + rK * K_LD + sK * 8);
      copy16B(vtg + (size_t)dV * SEQ + nxt + sV * 8,
              Vlds + (cur ^ 1) * VCH_SZ + dV * VT_LD + sV * 8);
    }

    const _Float16* Kb = Klds + cur * KCH_SZ;
    const _Float16* Vb = Vlds + cur * VCH_SZ;

    // S = Q K^T : two 16x16 tiles covering kv columns [kvb, kvb+32)
    v8f s0 = {}, s1 = {};
    {
      v16h b0 = frag_load(Kb, K_LD);
      s0 = wmma16(aq0, b0, s0);
      v16h b1 = frag_load(Kb + 32, K_LD);
      s0 = wmma16(aq1, b1, s0);
      v16h b2 = frag_load(Kb + 16 * K_LD, K_LD);
      s1 = wmma16(aq0, b2, s1);
      v16h b3 = frag_load(Kb + 16 * K_LD + 32, K_LD);
      s1 = wmma16(aq1, b3, s1);
    }

    // online softmax update (fp32), causal mask, write P to LDS as f16
#pragma unroll
    for (int j = 0; j < 8; ++j) {
      const int m   = j + 8 * hid;
      const int row = qbase + w * 16 + m;
      const bool ok0 = (kvb + n)      <= row;
      const bool ok1 = (kvb + 16 + n) <= row;
      const float sm0 = ok0 ? s0[j] : -3.0e38f;
      const float sm1 = ok1 ? s1[j] : -3.0e38f;

      float mx = fmaxf(sm0, sm1);
      mx = fmaxf(mx, __shfl_xor(mx, 1, 32));
      mx = fmaxf(mx, __shfl_xor(mx, 2, 32));
      mx = fmaxf(mx, __shfl_xor(mx, 4, 32));
      mx = fmaxf(mx, __shfl_xor(mx, 8, 32));

      const float mtot = fmaxf(m_run[j], mx);
      const float sc   = __expf(m_run[j] - mtot);
      const float p0   = ok0 ? __expf(s0[j] - mtot) : 0.0f;
      const float p1   = ok1 ? __expf(s1[j] - mtot) : 0.0f;

      float rs = p0 + p1;
      rs += __shfl_xor(rs, 1, 32);
      rs += __shfl_xor(rs, 2, 32);
      rs += __shfl_xor(rs, 4, 32);
      rs += __shfl_xor(rs, 8, 32);

      m_run[j] = mtot;
      l_run[j] = l_run[j] * sc + rs;
#pragma unroll
      for (int t = 0; t < 4; ++t) acc[t][j] *= sc;

      myP[m * 32 + n]      = (_Float16)p0;
      myP[m * 32 + 16 + n] = (_Float16)p1;
    }

    // in-wave LDS RAW: drain DS before reading other lanes' P writes
    asm volatile("s_wait_dscnt 0x0" ::: "memory");

    // ctx += P (16x32) * V (32x64): 4 WMMAs via pre-transposed V in LDS
    {
      const v16h aP = frag_load(myP, 32);
#pragma unroll
      for (int t = 0; t < 4; ++t) {
        v16h bV = frag_load(Vb + (size_t)(t * 16) * VT_LD, VT_LD);
        acc[t] = wmma16(aP, bV, acc[t]);
      }
    }
    cur ^= 1;
  }

  // normalize and store ctx (f16)
#pragma unroll
  for (int j = 0; j < 8; ++j) {
    const int m   = j + 8 * hid;
    const int row = qbase + w * 16 + m;
    const float inv_l = 1.0f / l_run[j];
#pragma unroll
    for (int t = 0; t < 4; ++t) {
      ctx_h[bo64 + (size_t)row * DK + t * 16 + n] =
          (_Float16)(acc[t][j] * inv_l);
    }
  }
}

// ---------------------------------------------------------------------------
// k3: output projection  out[m][h] = sum_v ctx[m][v] * Wo[h][v] + bo[h]
// ---------------------------------------------------------------------------
__global__ void k3_proj(const _Float16* __restrict__ ctx_h,
                        const _Float16* __restrict__ Wo_h,
                        const float* __restrict__ bo,
                        float* __restrict__ out) {
  const int tid  = threadIdx.x;
  const int w    = tid >> 5;
  const int lane = tid & 31;
  const int hid  = lane >> 4;
  const int n    = lane & 15;

  const int rbase = blockIdx.x * 128 + w * 16;
  const _Float16* ap = ctx_h + (size_t)rbase * DK;
  const v16h a0 = frag_load(ap, DK);
  const v16h a1 = frag_load(ap + 32, DK);

  for (int nt = 0; nt < HID / 16; ++nt) {
    v8f acc = {};
    v16h b0 = frag_load(Wo_h + (size_t)(nt * 16) * DK, DK);
    acc = wmma16(a0, b0, acc);
    v16h b1 = frag_load(Wo_h + (size_t)(nt * 16) * DK + 32, DK);
    acc = wmma16(a1, b1, acc);

    const float bval = bo[nt * 16 + n];
#pragma unroll
    for (int j = 0; j < 8; ++j) {
      const int m = j + 8 * hid;
      out[(size_t)(rbase + m) * HID + nt * 16 + n] = acc[j] + bval;
    }
  }
}

// ---------------------------------------------------------------------------
extern "C" void kernel_launch(void* const* d_in, const int* in_sizes, int n_in,
                              void* d_out, int out_size, void* d_ws, size_t ws_size,
                              hipStream_t stream) {
  (void)in_sizes; (void)n_in; (void)out_size; (void)ws_size;

  const float* y  = (const float*)d_in[0];
  const float* Wq = (const float*)d_in[1];
  const float* bq = (const float*)d_in[2];
  const float* Wk = (const float*)d_in[3];
  const float* bk = (const float*)d_in[4];
  const float* Wv = (const float*)d_in[5];
  const float* bv = (const float*)d_in[6];
  const float* Wo = (const float*)d_in[7];
  const float* bo = (const float*)d_in[8];
  float* out = (float*)d_out;

  // workspace layout (f16)
  char* ws = (char*)d_ws;
  _Float16* Wqkv_h = (_Float16*)(ws);                        // 192*1024  (384 KB)
  _Float16* Wo_h   = (_Float16*)(ws + 393216);               // 1024*64   (128 KB)
  _Float16* q_h    = (_Float16*)(ws + 524288);               // [b][t][d] (2 MB)
  _Float16* k_h    = (_Float16*)(ws + 524288 + 2097152);     // [b][t][d] (2 MB)
  _Float16* v_t    = (_Float16*)(ws + 524288 + 2 * 2097152); // [b][d][t] (2 MB)
  _Float16* ctx_h  = (_Float16*)(ws + 524288 + 3 * 2097152); // [b][t][d] (2 MB)

  k0_convert_w<<<1024, 256, 0, stream>>>(Wq, Wk, Wv, Wo, Wqkv_h, Wo_h);
  k1_qkv<<<NROWS / 16, 384, 0, stream>>>(y, Wqkv_h, bq, bk, bv, q_h, k_h, v_t);
  k2_attn<<<NBAT * (SEQ / QTILE), 256, 0, stream>>>(q_h, k_h, v_t, ctx_h);
  k3_proj<<<NROWS / 128, 256, 0, stream>>>(ctx_h, Wo_h, bo, out);
}
